// MagneticEdgeConvHiddenState_17935783428362
// MI455X (gfx1250) — compile-verified, matrix operand order checked by reading, not run on hardware
//
#include <hip/hip_runtime.h>

typedef __attribute__((ext_vector_type(16))) _Float16 v16h;
typedef __attribute__((ext_vector_type(8)))  _Float16 v8h;
typedef __attribute__((ext_vector_type(8)))  float    v8f;

#define NEDGES 800000
#define NNODES 50000
#define INCH   128
#define HID    64
#define OUTCH  128
#define S45    0.70710678118654752f

// LDS row strides (halves): +8 padding so strided B-frag reads spread banks
#define STRIDE128 136   // for 128-wide weight rows (Win, Wskip)
#define STRIDE64  72    // for 64-wide weight rows (Wout)

// ---- fragment loaders ------------------------------------------------------
// A-frag (16x32 f16) per-lane layout: row = lane&15, kb = (lane>>4)*8,
// halves 0..7  -> K = kc+kb+0..7, halves 8..15 -> K = kc+16+kb+0..7.
static __device__ inline v16h load_fragA_f32(const float* __restrict__ rowp, int kb) {
  float4 a = *(const float4*)(rowp + kb);
  float4 b = *(const float4*)(rowp + kb + 4);
  float4 c = *(const float4*)(rowp + kb + 16);
  float4 d = *(const float4*)(rowp + kb + 20);
  v16h r;
  r[0]=(_Float16)a.x; r[1]=(_Float16)a.y; r[2]=(_Float16)a.z;  r[3]=(_Float16)a.w;
  r[4]=(_Float16)b.x; r[5]=(_Float16)b.y; r[6]=(_Float16)b.z;  r[7]=(_Float16)b.w;
  r[8]=(_Float16)c.x; r[9]=(_Float16)c.y; r[10]=(_Float16)c.z; r[11]=(_Float16)c.w;
  r[12]=(_Float16)d.x; r[13]=(_Float16)d.y; r[14]=(_Float16)d.z; r[15]=(_Float16)d.w;
  return r;
}

// B-frag from LDS: same per-lane pattern over row-major f16 weight row (X @ W^T)
static __device__ inline v16h load_fragB_lds(const _Float16* rowp, int kb) {
  v8h lo = *(const v8h*)(rowp + kb);
  v8h hi = *(const v8h*)(rowp + kb + 16);
  return __builtin_shufflevector(lo, hi, 0,1,2,3,4,5,6,7,8,9,10,11,12,13,14,15);
}

#define WMMA(A,B,C) __builtin_amdgcn_wmma_f32_16x16x32_f16(false,(A),false,(B),(short)0,(C),false,false)

// ---- K0: zero node state h -------------------------------------------------
__global__ void __launch_bounds__(256)
k0_zero(float* __restrict__ h) {
  int i = blockIdx.x * blockDim.x + threadIdx.x;
  if (i < NNODES * HID) h[i] = 0.0f;
}

// ---- K1: z = (attr @ Win^T) * dinv ; complex scatter to h (atomics) --------
__global__ void __launch_bounds__(256)
k1_proj_scatter(const float* __restrict__ attr, const int* __restrict__ ei,
                const unsigned char* __restrict__ und,
                const float* __restrict__ WinF, float* __restrict__ h) {
  __shared__ __align__(16) _Float16 sWin[HID * STRIDE128];      // 17,408 B
  for (int i = threadIdx.x; i < HID * INCH; i += 256) {
    int r = i >> 7, c = i & 127;
    sWin[r * STRIDE128 + c] = (_Float16)WinF[i];
  }
  __syncthreads();

  int wave  = (int)((blockIdx.x * blockDim.x + threadIdx.x) >> 5);
  int lane  = threadIdx.x & 31;
  int ebase = wave * 16;
  int row   = lane & 15;
  int kb    = (lane >> 4) << 3;
  int e     = ebase + row;
  int u0 = ei[e];
  int v0 = ei[NEDGES + e];
  int q0 = und[e];

  const float* arow = attr + (size_t)e * INCH;
  v16h af0 = load_fragA_f32(arow +  0, kb);
  v16h af1 = load_fragA_f32(arow + 32, kb);
  v16h af2 = load_fragA_f32(arow + 64, kb);
  v16h af3 = load_fragA_f32(arow + 96, kb);

  v8f acc[4];
  #pragma unroll
  for (int t = 0; t < 4; ++t) {
    const _Float16* wr = sWin + (size_t)(t * 16 + row) * STRIDE128;
    v8f c = {};
    c = WMMA(af0, load_fragB_lds(wr +  0, kb), c);
    c = WMMA(af1, load_fragB_lds(wr + 32, kb), c);
    c = WMMA(af2, load_fragB_lds(wr + 64, kb), c);
    c = WMMA(af3, load_fragB_lds(wr + 96, kb), c);
    acc[t] = c;
  }

  // D layout: n = lane&15 (per 16-col tile), m = r + 8*(lane>>4)
  int nlo   = lane & 15;
  float sgn = (nlo & 1) ? 1.0f : -1.0f;  // even slot = real (needs -b*imag)
  #pragma unroll
  for (int r = 0; r < 8; ++r) {
    int m  = r + kb;                 // kb == 8*(lane>>4)
    int um = __shfl(u0, m, 32);
    int vm = __shfl(v0, m, 32);
    int qm = __shfl(q0, m, 32);
    float dinv = (um == vm && qm) ? 0.0f : S45;
    // c_head = conj(phase); c_tail = -phase; phase = und ? 1 : exp(i*pi/4)
    float ah = qm ?  1.0f :  S45;
    float bh = qm ?  0.0f : -S45;
    float at = qm ? -1.0f : -S45;
    float bt = qm ?  0.0f : -S45;
    #pragma unroll
    for (int t = 0; t < 4; ++t) {
      float val = acc[t][r] * dinv;
      float par = __shfl_xor(val, 1, 32);   // partner component (lane^1, same m)
      int n = t * 16 + nlo;
      float ho = ah * val + sgn * bh * par;
      float to = at * val + sgn * bt * par;
      atomicAdd(h + (size_t)vm * HID + n, ho);
      atomicAdd(h + (size_t)um * HID + n, to);
    }
  }
}

// ---- K2: ReLU on node state ------------------------------------------------
__global__ void __launch_bounds__(256)
k2_relu(float* __restrict__ h) {
  int i = blockIdx.x * blockDim.x + threadIdx.x;
  if (i < NNODES * HID) h[i] = fmaxf(h[i], 0.0f);
}

// ---- K3: gather, complex combine, out = y@Wout^T + attr@Wskip^T + biases ---
__global__ void __launch_bounds__(256)
k3_gather_out(const float* __restrict__ attr, const int* __restrict__ ei,
              const unsigned char* __restrict__ und,
              const float* __restrict__ WoutF, const float* __restrict__ WskipF,
              const float* __restrict__ bskip, const float* __restrict__ bias,
              const float* __restrict__ h, float* __restrict__ out) {
  __shared__ __align__(16) _Float16 sWout[OUTCH * STRIDE64];    // 18,432 B
  __shared__ __align__(16) _Float16 sWskip[OUTCH * STRIDE128];  // 34,816 B
  for (int i = threadIdx.x; i < OUTCH * HID; i += 256) {
    int r = i >> 6, c = i & 63;
    sWout[r * STRIDE64 + c] = (_Float16)WoutF[i];
  }
  for (int i = threadIdx.x; i < OUTCH * INCH; i += 256) {
    int r = i >> 7, c = i & 127;
    sWskip[r * STRIDE128 + c] = (_Float16)WskipF[i];
  }
  __syncthreads();

  int wave  = (int)((blockIdx.x * blockDim.x + threadIdx.x) >> 5);
  int lane  = threadIdx.x & 31;
  int ebase = wave * 16;
  int row   = lane & 15;
  int kb    = (lane >> 4) << 3;
  int e     = ebase + row;
  int u0 = ei[e];
  int v0 = ei[NEDGES + e];
  int q  = und[e];
  float dinv = (u0 == v0 && q) ? 0.0f : S45;
  // conj(c_head) = phase ; conj(c_tail) = -conj(phase)
  float ga = q ?  1.0f :  S45;
  float gb = q ?  0.0f :  S45;
  float ta = q ? -1.0f : -S45;
  float tb = q ?  0.0f :  S45;
  const float* hv = h + (size_t)v0 * HID;
  const float* hu = h + (size_t)u0 * HID;

  // y A-fragments (16x64 -> 2 chunks of K=32); runs of 8 are 4 complex pairs
  v16h yf[2];
  #pragma unroll
  for (int kc2 = 0; kc2 < 2; ++kc2) {
    int kc = kc2 * 32;
    v16h r;
    #pragma unroll
    for (int half = 0; half < 2; ++half) {
      int base = kc + half * 16 + kb;   // always even: base+2p real, +2p+1 imag
      float4 a0 = *(const float4*)(hv + base);
      float4 a1 = *(const float4*)(hv + base + 4);
      float4 b0 = *(const float4*)(hu + base);
      float4 b1 = *(const float4*)(hu + base + 4);
      int o = half * 8;
      r[o+0] = (_Float16)(dinv * (ga*a0.x - gb*a0.y + ta*b0.x - tb*b0.y));
      r[o+1] = (_Float16)(dinv * (ga*a0.y + gb*a0.x + ta*b0.y + tb*b0.x));
      r[o+2] = (_Float16)(dinv * (ga*a0.z - gb*a0.w + ta*b0.z - tb*b0.w));
      r[o+3] = (_Float16)(dinv * (ga*a0.w + gb*a0.z + ta*b0.w + tb*b0.z));
      r[o+4] = (_Float16)(dinv * (ga*a1.x - gb*a1.y + ta*b1.x - tb*b1.y));
      r[o+5] = (_Float16)(dinv * (ga*a1.y + gb*a1.x + ta*b1.y + tb*b1.x));
      r[o+6] = (_Float16)(dinv * (ga*a1.z - gb*a1.w + ta*b1.z - tb*b1.w));
      r[o+7] = (_Float16)(dinv * (ga*a1.w + gb*a1.z + ta*b1.w + tb*b1.z));
    }
    yf[kc2] = r;
  }

  const float* arow = attr + (size_t)e * INCH;
  v16h af[4];
  #pragma unroll
  for (int kc = 0; kc < 4; ++kc) af[kc] = load_fragA_f32(arow + kc * 32, kb);

  int nlo = lane & 15;
  #pragma unroll
  for (int t = 0; t < 8; ++t) {
    v8f c = {};
    const _Float16* wo = sWout + (size_t)(t * 16 + row) * STRIDE64;
    c = WMMA(yf[0], load_fragB_lds(wo +  0, kb), c);
    c = WMMA(yf[1], load_fragB_lds(wo + 32, kb), c);
    const _Float16* wk = sWskip + (size_t)(t * 16 + row) * STRIDE128;
    #pragma unroll
    for (int kc = 0; kc < 4; ++kc)
      c = WMMA(af[kc], load_fragB_lds(wk + kc * 32, kb), c);

    int n = t * 16 + nlo;
    float bsum = bskip[n] + bias[n];
    float* orow = out + (size_t)(ebase + kb) * OUTCH + n;  // m = r + kb
    #pragma unroll
    for (int r2 = 0; r2 < 8; ++r2)
      orow[(size_t)r2 * OUTCH] = c[r2] + bsum;
  }
}

// ---- launch ---------------------------------------------------------------
extern "C" void kernel_launch(void* const* d_in, const int* in_sizes, int n_in,
                              void* d_out, int out_size, void* d_ws, size_t ws_size,
                              hipStream_t stream) {
  (void)in_sizes; (void)n_in; (void)out_size; (void)ws_size;
  const int*           ei     = (const int*)d_in[0];
  const unsigned char* und    = (const unsigned char*)d_in[1];  // jax bool = 1 byte
  const float*         attr   = (const float*)d_in[2];
  const float*         WinF   = (const float*)d_in[3];
  const float*         WoutF  = (const float*)d_in[4];
  const float*         WskipF = (const float*)d_in[5];
  const float*         bskip  = (const float*)d_in[6];
  const float*         bias   = (const float*)d_in[7];
  float*               out    = (float*)d_out;

  float* h = (float*)d_ws;   // 12,800,000 B node state, L2-resident

  // 3.2M threads exactly cover h; 50,000 waves exactly cover 800,000/16 tiles
  k0_zero        <<<12500, 256, 0, stream>>>(h);
  k1_proj_scatter<<< 6250, 256, 0, stream>>>(attr, ei, und, WinF, h);
  k2_relu        <<<12500, 256, 0, stream>>>(h);
  k3_gather_out  <<< 6250, 256, 0, stream>>>(attr, ei, und, WoutF, WskipF, bskip, bias, h, out);
}
